// AttentionBlock_20581483282632
// MI455X (gfx1250) — matrix operand order repaired and checked by reading, not compiled, not run on hardware
//
#include <hip/hip_runtime.h>
#include <hip/hip_bf16.h>
#include <math.h>

// AttentionBlock for MI455X (gfx1250, wave32, WMMA).
// B=32, C=512, T=1024, KEY=VALUE=512.
// Compute-bound (~120 GFLOP vs ~130MB traffic @ 23.3TB/s) -> all GEMMs on
// v_wmma_f32_16x16x32_bf16 with 2D register blocking (8 accumulators/wave,
// both operand fragments reused) to maximize WMMA per operand load.
// Workspace (bf16): qb(32MB) | kb(32MB) | vb(32MB) | scores(64MB) = 160MB.

#define BB 32
#define CC 512
#define TT 1024
#define KK 512   // KEY_SIZE == VALUE_SIZE

typedef __attribute__((ext_vector_type(16))) __bf16 v16bf;
typedef __attribute__((ext_vector_type(8)))  float  v8f;

union Frag16 {
    v16bf v;
    uint4 q[2];
    unsigned short s[16];
};

static __device__ __forceinline__ unsigned short bf16_bits(float f) {
    union { __bf16 h; unsigned short u; } v;
    v.h = (__bf16)f;                    // native v_cvt_pk_bf16_f32 path
    return v.u;
}

static __device__ __forceinline__ float bf16_to_f32(unsigned short u) {
    union { unsigned int u; float f; } v; v.u = ((unsigned int)u) << 16;
    return v.f;
}

static __device__ __forceinline__ v8f wmma_bf16(const Frag16& a, const Frag16& b, v8f c) {
    return __builtin_amdgcn_wmma_f32_16x16x32_bf16(
        /*neg_a=*/false, a.v, /*neg_b=*/false, b.v,
        /*c_mod=*/(short)0, c, /*reuse_a=*/false, /*reuse_b=*/false);
}

// ---------------------------------------------------------------------------
// Kernel 1: q/k/v = xt @ W^T + b   (x is [B,C,T]; xt[t,c] = x[b,c,t]).
// Wave computes 32(t) x 64(n): one 32x32 x-tile transposed through LDS per
// K-chunk; 2 A-fragments x 4 W-fragments -> 8 WMMAs per chunk.
// wid = which*2^13 + b*2^8 + t_blk*2^3 + n_blk
// ---------------------------------------------------------------------------
__global__ __launch_bounds__(256) void qkv_proj_kernel(
    const float* __restrict__ x,
    const float* __restrict__ Wq, const float* __restrict__ bq,
    const float* __restrict__ Wk, const float* __restrict__ bk,
    const float* __restrict__ Wv, const float* __restrict__ bv,
    unsigned short* __restrict__ qb,
    unsigned short* __restrict__ kb,
    unsigned short* __restrict__ vb)
{
    __shared__ __align__(16) unsigned short As[8][32 * 32];   // per-wave 2KB slice

    const int lane  = threadIdx.x & 31;
    const int wslot = threadIdx.x >> 5;
    const unsigned wid = blockIdx.x * 8u + wslot;

    const int n_blk = wid & 7;                    // 8 blocks of 64 features
    const int t_blk = (wid >> 3) & 31;            // 32 blocks of 32 queries
    const int b     = (wid >> 8) & 31;
    const int which = (int)(wid >> 13);           // 0=q, 1=k, 2=v

    const float* W    = (which == 0) ? Wq : (which == 1) ? Wk : Wv;
    const float* bias = (which == 0) ? bq : (which == 1) ? bk : bv;
    unsigned short* out = (which == 0) ? qb : (which == 1) ? kb : vb;

    const int h   = lane >> 4;
    const int l15 = lane & 15;
    const int t0  = t_blk * 32;
    const int n0  = n_blk * 64;

    unsigned short* as = As[wslot];
    v8f acc[2][4] = {};

    for (int c0 = 0; c0 < CC; c0 += 32) {
        __syncthreads();   // guard previous iteration's LDS reads
        // Stage A: lane owns channel c = c0+lane, loads 32 contiguous t values.
        const float* xp = x + ((size_t)(b * CC + c0 + lane)) * TT + t0;
        __builtin_prefetch(xp + 32 * TT, 0, 0);   // next K-chunk
        float tf[32];
        #pragma unroll
        for (int j = 0; j < 8; ++j)
            *(float4*)&tf[4 * j] = ((const float4*)xp)[j];
        #pragma unroll
        for (int j = 0; j < 32; ++j)
            as[j * 32 + lane] = bf16_bits(tf[j]);   // transposed: As[t][c]
        __syncthreads();

        // 2 A fragments: rows m = tt*16 + l15, k-chunks {8h..} and {16+8h..}.
        Frag16 a[2];
        #pragma unroll
        for (int tt = 0; tt < 2; ++tt) {
            a[tt].q[0] = *(const uint4*)(as + (tt * 16 + l15) * 32 + 8 * h);
            a[tt].q[1] = *(const uint4*)(as + (tt * 16 + l15) * 32 + 16 + 8 * h);
        }

        // 4 W fragments (B^T rows are contiguous); each feeds 2 WMMAs.
        #pragma unroll
        for (int u = 0; u < 4; ++u) {
            const float* wp = W + (size_t)(n0 + u * 16 + l15) * CC + c0 + 8 * h;
            float wf[16];
            *(float4*)&wf[0]  = ((const float4*)wp)[0];
            *(float4*)&wf[4]  = ((const float4*)wp)[1];
            *(float4*)&wf[8]  = ((const float4*)wp)[4];
            *(float4*)&wf[12] = ((const float4*)wp)[5];
            Frag16 bf;
            #pragma unroll
            for (int j = 0; j < 16; ++j) bf.v[j] = (__bf16)wf[j];
            #pragma unroll
            for (int tt = 0; tt < 2; ++tt)
                acc[tt][u] = wmma_bf16(a[tt], bf, acc[tt][u]);
        }
    }

    // C/D layout: N = l15, M = vv + 8h. Add bias, store bf16.
    #pragma unroll
    for (int tt = 0; tt < 2; ++tt) {
        #pragma unroll
        for (int u = 0; u < 4; ++u) {
            const int n = n0 + u * 16 + l15;
            const float bn = bias[n];
            unsigned short* op =
                out + ((size_t)b * TT + t0 + tt * 16 + 8 * h) * KK + n;
            #pragma unroll
            for (int vv = 0; vv < 8; ++vv)
                op[(size_t)vv * KK] = bf16_bits(acc[tt][u][vv] + bn);
        }
    }
}

// ---------------------------------------------------------------------------
// Kernel 2: scores[b,t,s] = (q[b,t,:] . k[b,s,:]) / sqrt(512), causal mask s>t.
// Wave computes 32(t) x 64(s): 12 b128 loads -> 8 WMMAs per K-chunk.
// wid = b*2^9 + t_blk*2^4 + s_blk
// ---------------------------------------------------------------------------
__global__ __launch_bounds__(256) void scores_kernel(
    const unsigned short* __restrict__ qb,
    const unsigned short* __restrict__ kb,
    unsigned short* __restrict__ sb)
{
    const int lane  = threadIdx.x & 31;
    const int wslot = threadIdx.x >> 5;
    const unsigned wid = blockIdx.x * 8u + wslot;

    const int s_blk = wid & 15;                   // 16 blocks of 64 keys
    const int t_blk = (wid >> 4) & 31;            // 32 blocks of 32 queries
    const int b     = (int)(wid >> 9);

    const int h   = lane >> 4;
    const int l15 = lane & 15;
    const int t0  = t_blk * 32;
    const int s0  = s_blk * 64;

    const unsigned short* qr0 = qb + ((size_t)b * TT + t0 + l15) * KK;
    const unsigned short* kr0 = kb + ((size_t)b * TT + s0 + l15) * KK;

    v8f acc[2][4] = {};
    for (int k0 = 0; k0 < KK; k0 += 32) {
        __builtin_prefetch(qr0 + k0 + 64, 0, 0);
        Frag16 a[2];
        #pragma unroll
        for (int tt = 0; tt < 2; ++tt) {
            const unsigned short* qrow = qr0 + (size_t)tt * 16 * KK;
            a[tt].q[0] = *(const uint4*)(qrow + k0 + 8 * h);
            a[tt].q[1] = *(const uint4*)(qrow + k0 + 16 + 8 * h);
        }
        #pragma unroll
        for (int u = 0; u < 4; ++u) {
            const unsigned short* krow = kr0 + (size_t)u * 16 * KK;
            Frag16 bf;
            bf.q[0] = *(const uint4*)(krow + k0 + 8 * h);
            bf.q[1] = *(const uint4*)(krow + k0 + 16 + 8 * h);
            #pragma unroll
            for (int tt = 0; tt < 2; ++tt)
                acc[tt][u] = wmma_bf16(a[tt], bf, acc[tt][u]);
        }
    }

    const float scale = 0.04419417382415922f;   // 1/sqrt(512)
    #pragma unroll
    for (int tt = 0; tt < 2; ++tt) {
        #pragma unroll
        for (int u = 0; u < 4; ++u) {
            const int s = s0 + u * 16 + l15;
            unsigned short* op =
                sb + ((size_t)b * TT + t0 + tt * 16 + 8 * h) * TT + s;
            #pragma unroll
            for (int vv = 0; vv < 8; ++vv) {
                const int t = t0 + tt * 16 + 8 * h + vv;
                op[(size_t)vv * TT] = (s > t) ? (unsigned short)0xFF80u  // bf16 -inf
                                              : bf16_bits(acc[tt][u][vv] * scale);
            }
        }
    }
}

// ---------------------------------------------------------------------------
// Kernel 3: softmax over axis=1 (the QUERY axis t) per (b, s) column.
// 64 columns per block -> coalesced row reads; 4 threads/column reduce in LDS.
// grid = B * (T/64)
// ---------------------------------------------------------------------------
__global__ __launch_bounds__(256) void softmax_col_kernel(unsigned short* __restrict__ sb)
{
    const int b  = blockIdx.x >> 4;
    const int s0 = (blockIdx.x & 15) * 64;
    const int col = threadIdx.x & 63;
    const int yy  = threadIdx.x >> 6;   // 0..3

    unsigned short* base = sb + (size_t)b * TT * TT + s0 + col;

    __shared__ float red[4][64];

    float m = -INFINITY;
    for (int t = yy; t < TT; t += 4)
        m = fmaxf(m, bf16_to_f32(base[(size_t)t * TT]));
    red[yy][col] = m;
    __syncthreads();
    const float cm = fmaxf(fmaxf(red[0][col], red[1][col]),
                           fmaxf(red[2][col], red[3][col]));
    __syncthreads();

    float ss = 0.0f;
    for (int t = yy; t < TT; t += 4)
        ss += __expf(bf16_to_f32(base[(size_t)t * TT]) - cm);
    red[yy][col] = ss;
    __syncthreads();
    const float inv = 1.0f / (red[0][col] + red[1][col] + red[2][col] + red[3][col]);

    for (int t = yy; t < TT; t += 4) {
        const float w = __expf(bf16_to_f32(base[(size_t)t * TT]) - cm) * inv;
        base[(size_t)t * TT] = bf16_bits(w);
    }
}

// ---------------------------------------------------------------------------
// Kernel 4: out[b,n,t] = sum_s w[b,t,s] * v[b,s,n]   (output already transposed).
// Wave computes 64(t) x 32(n): one 32x32 v-tile transposed through LDS per
// K-chunk; 4 A-fragments x 2 B-fragments -> 8 WMMAs per chunk.
// wid = b*2^8 + t_blk*2^4 + n_blk
// ---------------------------------------------------------------------------
__global__ __launch_bounds__(256) void out_gemm_kernel(
    const unsigned short* __restrict__ sb,
    const unsigned short* __restrict__ vbuf,
    float* __restrict__ out)
{
    __shared__ __align__(16) unsigned short Vs[8][32 * 32];   // per-wave 2KB slice

    const int lane  = threadIdx.x & 31;
    const int wslot = threadIdx.x >> 5;
    const unsigned wid = blockIdx.x * 8u + wslot;

    const int n_blk = wid & 15;                   // 16 blocks of 32 features
    const int t_blk = (wid >> 4) & 15;            // 16 blocks of 64 queries
    const int b     = (int)(wid >> 8);

    const int h   = lane >> 4;
    const int l15 = lane & 15;
    const int t0  = t_blk * 64;
    const int n0  = n_blk * 32;

    unsigned short* vs = Vs[wslot];
    v8f acc[4][2] = {};

    for (int ss0 = 0; ss0 < TT; ss0 += 32) {
        // Stage v tile transposed: lane owns row s = ss0+lane, 32 cols from n0.
        const unsigned short* vp = vbuf + ((size_t)b * TT + ss0 + lane) * KK + n0;
        unsigned short tmp[32];
        #pragma unroll
        for (int j = 0; j < 4; ++j)
            *(uint4*)&tmp[8 * j] = ((const uint4*)vp)[j];
        __syncthreads();   // guard previous iteration's LDS reads
        #pragma unroll
        for (int j = 0; j < 32; ++j)
            vs[j * 32 + lane] = tmp[j];    // Vs[n][s]
        __syncthreads();

        // 2 B fragments: column n = u*16 + l15, s values at {8h..} and {16+8h..}. Each x4 reuse.
        Frag16 bf[2];
        #pragma unroll
        for (int u = 0; u < 2; ++u) {
            bf[u].q[0] = *(const uint4*)(vs + (u * 16 + l15) * 32 + 8 * h);
            bf[u].q[1] = *(const uint4*)(vs + (u * 16 + l15) * 32 + 16 + 8 * h);
        }

        #pragma unroll
        for (int ut = 0; ut < 4; ++ut) {
            const unsigned short* wrow =
                sb + ((size_t)b * TT + t0 + ut * 16 + l15) * TT;
            __builtin_prefetch(wrow + ss0 + 64, 0, 0);
            Frag16 a;
            a.q[0] = *(const uint4*)(wrow + ss0 + 8 * h);
            a.q[1] = *(const uint4*)(wrow + ss0 + 16 + 8 * h);
            #pragma unroll
            for (int u = 0; u < 2; ++u)
                acc[ut][u] = wmma_bf16(a, bf[u], acc[ut][u]);
        }
    }

    // out[b][n][t]: n = n0 + u*16 + l15 per lane; t = t0 + ut*16 + 8h + vv.
    #pragma unroll
    for (int ut = 0; ut < 4; ++ut) {
        #pragma unroll
        for (int u = 0; u < 2; ++u) {
            float* op = out + ((size_t)b * KK + n0 + u * 16 + l15) * TT
                            + t0 + ut * 16 + 8 * h;
            float4 o0 = make_float4(acc[ut][u][0], acc[ut][u][1],
                                    acc[ut][u][2], acc[ut][u][3]);
            float4 o1 = make_float4(acc[ut][u][4], acc[ut][u][5],
                                    acc[ut][u][6], acc[ut][u][7]);
            ((float4*)op)[0] = o0;
            ((float4*)op)[1] = o1;
        }
    }
}

// ---------------------------------------------------------------------------
extern "C" void kernel_launch(void* const* d_in, const int* in_sizes, int n_in,
                              void* d_out, int out_size, void* d_ws, size_t ws_size,
                              hipStream_t stream)
{
    const float* x  = (const float*)d_in[0];
    const float* Wq = (const float*)d_in[1];
    const float* bq = (const float*)d_in[2];
    const float* Wk = (const float*)d_in[3];
    const float* bk = (const float*)d_in[4];
    const float* Wv = (const float*)d_in[5];
    const float* bv = (const float*)d_in[6];
    float* out = (float*)d_out;

    // Workspace layout (bf16 elements): qb | kb | vb | scores  = 160 MB total.
    const size_t QKV_ELEMS = (size_t)BB * TT * KK;       // 16M each
    unsigned short* qb = (unsigned short*)d_ws;
    unsigned short* kb = qb + QKV_ELEMS;
    unsigned short* vb = kb + QKV_ELEMS;
    unsigned short* sb = vb + QKV_ELEMS;
    (void)ws_size; (void)in_sizes; (void)n_in; (void)out_size;

    // 1) projections: 3 * 32 * 32 * 8 = 24576 waves / 8 per block
    qkv_proj_kernel<<<3072, 256, 0, stream>>>(x, Wq, bq, Wk, bk, Wv, bv, qb, kb, vb);
    // 2) scores: 32 * 32 * 16 = 16384 waves
    scores_kernel<<<2048, 256, 0, stream>>>(qb, kb, sb);
    // 3) column softmax: 32 * 16 blocks
    softmax_col_kernel<<<512, 256, 0, stream>>>(sb);
    // 4) output GEMM (+transpose): 32 * 16 * 16 = 8192 waves
    out_gemm_kernel<<<1024, 256, 0, stream>>>(sb, vb, out);
}
